// AttentionBlock_41695542509821
// MI455X (gfx1250) — matrix operand order repaired and checked by reading, not compiled, not run on hardware
//
#include <hip/hip_runtime.h>
#include <hip/hip_bf16.h>

typedef __attribute__((ext_vector_type(8)))  _Float16 v8h;
typedef __attribute__((ext_vector_type(16))) _Float16 v16h;
typedef __attribute__((ext_vector_type(8)))  float    v8f;

#define C_DIM     512
#define N_SPATIAL 4096
#define BATCH     4
#define NTOT      (BATCH * N_SPATIAL)
#define QK_SCALE  0.04419417382415922f   // 512^-0.5
#define GN_EPS    1e-6f

// ---- WMMA helpers -----------------------------------------------------------
// A fragment (16x32 f16): lane<16 -> row=lane, K = {kA..kA+7, 16+kA..16+kA+7}, kA=0
//                         lane>=16 -> row=lane-16, kA=8
// -> two contiguous 16B loads at (row,kb+kA) and (row,kb+16+kA).
// B fragment (32x16 f16): same K order as A, col=lane%16
// -> two contiguous 16B loads at (col,kb+kA) and (col,kb+16+kA).
static __device__ __forceinline__ v16h load_frag(const _Float16* p0, const _Float16* p1) {
  v8h a = *(const v8h*)p0;
  v8h b = *(const v8h*)p1;
  v16h r;
#pragma unroll
  for (int i = 0; i < 8; ++i) { r[i] = a[i]; r[i + 8] = b[i]; }
  return r;
}

static __device__ __forceinline__ v8f wmma16(v16h a, v16h b, v8f c) {
  return __builtin_amdgcn_wmma_f32_16x16x32_f16(false, a, false, b, (short)0, c,
                                                false, false);
}

// ---- 1) GroupNorm: x fp32 [B,4096,512] -> h f16 -----------------------------
__global__ __launch_bounds__(256) void gn_kernel(const float* __restrict__ x,
                                                 const float* __restrict__ gamma,
                                                 const float* __restrict__ beta,
                                                 _Float16* __restrict__ h) {
  const int b = blockIdx.x >> 5;
  const int g = blockIdx.x & 31;
  const int tid = threadIdx.x;
  const int wave = tid >> 5, lane = tid & 31;

  __shared__ float redS[8], redQ[8];
  __shared__ float sg[16], sbt[16], stats[2];
  if (tid < 16) {
    sg[tid]  = gamma[g * 16 + tid];
    sbt[tid] = beta[g * 16 + tid];
  }

  const float* xb = x + (size_t)b * N_SPATIAL * C_DIM + g * 16;
  float sum = 0.f, sq = 0.f;
  for (int n = tid; n < N_SPATIAL; n += 256) {
    const float4* px = (const float4*)(xb + (size_t)n * C_DIM);
#pragma unroll
    for (int j = 0; j < 4; ++j) {
      float4 c = px[j];
      sum += c.x + c.y + c.z + c.w;
      sq  += c.x * c.x + c.y * c.y + c.z * c.z + c.w * c.w;
    }
  }
#pragma unroll
  for (int d = 16; d >= 1; d >>= 1) {
    sum += __shfl_xor(sum, d, 32);
    sq  += __shfl_xor(sq, d, 32);
  }
  if (lane == 0) { redS[wave] = sum; redQ[wave] = sq; }
  __syncthreads();
  if (tid == 0) {
    float s = 0.f, s2 = 0.f;
#pragma unroll
    for (int p = 0; p < 8; ++p) { s += redS[p]; s2 += redQ[p]; }
    const float inv = 1.0f / (float)(N_SPATIAL * 16);
    float mean = s * inv;
    float var  = s2 * inv - mean * mean;
    stats[0] = mean;
    stats[1] = rsqrtf(var + GN_EPS);
  }
  __syncthreads();
  const float mean = stats[0], rstd = stats[1];

  for (int n = tid; n < N_SPATIAL; n += 256) {
    const float* pr = xb + (size_t)n * C_DIM;
    v8h o0, o1;
#pragma unroll
    for (int c = 0; c < 8; ++c)
      o0[c] = (_Float16)((pr[c] - mean) * rstd * sg[c] + sbt[c]);
#pragma unroll
    for (int c = 0; c < 8; ++c)
      o1[c] = (_Float16)((pr[c + 8] - mean) * rstd * sg[c + 8] + sbt[c + 8]);
    _Float16* hr = h + (size_t)(b * N_SPATIAL + n) * C_DIM + g * 16;
    ((v8h*)hr)[0] = o0;
    ((v8h*)hr)[1] = o1;
  }
}

// ---- 2) Weight convert+transpose: w fp32 [K,N] -> wT f16 [N,K] --------------
__global__ __launch_bounds__(256) void wconv_kernel(const float* __restrict__ wq,
                                                    const float* __restrict__ wk,
                                                    const float* __restrict__ wv,
                                                    const float* __restrict__ wo,
                                                    _Float16* __restrict__ wT) {
  const int z = blockIdx.z;
  const float* w = (z == 0) ? wq : (z == 1) ? wk : (z == 2) ? wv : wo;
  _Float16* o = wT + (size_t)z * C_DIM * C_DIM;
  int idx = blockIdx.x * 256 + threadIdx.x;           // 262144 elements
  int n = idx >> 9, kk = idx & 511;
  o[(size_t)n * C_DIM + kk] = (_Float16)w[(size_t)kk * C_DIM + n];
}

// ---- 3) QKV GEMM: [16384,512] x [512,512] (+bias, q scaled, V transposed) ---
__global__ __launch_bounds__(128) void qkv_gemm_kernel(
    const _Float16* __restrict__ h, const _Float16* __restrict__ wT,
    const float* __restrict__ bq, const float* __restrict__ bk,
    const float* __restrict__ bv, _Float16* __restrict__ q,
    _Float16* __restrict__ k, _Float16* __restrict__ vT) {
  const int z = blockIdx.z;
  const _Float16* w   = wT + (size_t)z * C_DIM * C_DIM;
  const float* bias   = (z == 0) ? bq : (z == 1) ? bk : bv;
  const int wave = threadIdx.x >> 5, lane = threadIdx.x & 31;
  const int half = lane >> 4, li = lane & 15, kA = half * 8;
  const int m0 = blockIdx.x * 64 + (wave >> 1) * 32;
  const int n0 = blockIdx.y * 64 + (wave & 1) * 32;

  v8f zero = {};
  v8f acc[2][2];
#pragma unroll
  for (int i = 0; i < 2; ++i)
#pragma unroll
    for (int j = 0; j < 2; ++j) acc[i][j] = zero;

  for (int kb = 0; kb < C_DIM; kb += 32) {
    v16h a[2], bf[2];
#pragma unroll
    for (int i = 0; i < 2; ++i) {
      const _Float16* pa = h + (size_t)(m0 + i * 16 + li) * C_DIM + kb + kA;
      a[i] = load_frag(pa, pa + 16);
    }
#pragma unroll
    for (int j = 0; j < 2; ++j) {
      const _Float16* pb = w + (size_t)(n0 + j * 16 + li) * C_DIM + kb + kA;
      bf[j] = load_frag(pb, pb + 16);
    }
#pragma unroll
    for (int i = 0; i < 2; ++i)
#pragma unroll
      for (int j = 0; j < 2; ++j) acc[i][j] = wmma16(a[i], bf[j], acc[i][j]);
  }

  const float scale = (z == 0) ? QK_SCALE : 1.0f;
#pragma unroll
  for (int j = 0; j < 2; ++j) {
    const int col = n0 + j * 16 + li;
    const float bb = bias[col];
#pragma unroll
    for (int i = 0; i < 2; ++i) {
#pragma unroll
      for (int r = 0; r < 8; ++r) {
        const int row = m0 + i * 16 + r + 8 * half;
        const float val = (acc[i][j][r] + bb) * scale;
        if (z < 2) {
          _Float16* dst = (z == 0) ? q : k;
          dst[(size_t)row * C_DIM + col] = (_Float16)val;
        } else {  // V stored transposed per batch: vT[b][c][m]
          const int batch = row >> 12, ml = row & 4095;
          vT[(size_t)(batch * C_DIM + col) * N_SPATIAL + ml] = (_Float16)val;
        }
      }
    }
  }
}

// ---- 4) Flash attention: per block = 16 query rows, 8 waves split C ---------
__global__ __launch_bounds__(256) void attn_kernel(const _Float16* __restrict__ qm,
                                                   const _Float16* __restrict__ km,
                                                   const _Float16* __restrict__ vT,
                                                   _Float16* __restrict__ attn) {
  const int i0 = blockIdx.x * 16;
  const int b  = blockIdx.y;
  const int wave = threadIdx.x >> 5, lane = threadIdx.x & 31;
  const int half = lane >> 4, li = lane & 15, kA = half * 8;

  __shared__ float    Spar[8][16][32];   // per-wave partial score tiles
  __shared__ _Float16 Pl[16][32];        // probs (f16) for P@V
  __shared__ float    alphaS[16], mstate[16], lstate[16];

  if (threadIdx.x < 16) { mstate[threadIdx.x] = -1e30f; lstate[threadIdx.x] = 0.f; }

  // q fragments for this wave's 64-channel slice (reused for all key blocks)
  const _Float16* qb = qm + (size_t)(b * N_SPATIAL + i0 + li) * C_DIM + wave * 64 + kA;
  const v16h a0 = load_frag(qb, qb + 16);
  const v16h a1 = load_frag(qb + 32, qb + 48);

  v8f zero = {};
  v8f o[4];
#pragma unroll
  for (int t = 0; t < 4; ++t) o[t] = zero;
  __syncthreads();

  for (int j0 = 0; j0 < N_SPATIAL; j0 += 32) {
    // partial scores over this wave's C slice, 16 queries x 32 keys
    v8f s0 = zero, s1 = zero;
    {
      const _Float16* kp0 =
          km + (size_t)(b * N_SPATIAL + j0 + li) * C_DIM + wave * 64 + kA;
      s0 = wmma16(a0, load_frag(kp0, kp0 + 16), s0);
      s0 = wmma16(a1, load_frag(kp0 + 32, kp0 + 48), s0);
      const _Float16* kp1 = kp0 + (size_t)16 * C_DIM;
      s1 = wmma16(a0, load_frag(kp1, kp1 + 16), s1);
      s1 = wmma16(a1, load_frag(kp1 + 32, kp1 + 48), s1);
    }
#pragma unroll
    for (int r = 0; r < 8; ++r) {
      Spar[wave][r + 8 * half][li]      = s0[r];
      Spar[wave][r + 8 * half][li + 16] = s1[r];
    }
    __syncthreads();

    // online softmax: wave handles rows (2w, 2w+1) via its two lane halves
    {
      const int row = 2 * wave + half;
      float sA = 0.f, sB = 0.f;
#pragma unroll
      for (int p = 0; p < 8; ++p) {
        sA += Spar[p][row][li];
        sB += Spar[p][row][li + 16];
      }
      float mloc = fmaxf(sA, sB);
#pragma unroll
      for (int d = 8; d >= 1; d >>= 1) mloc = fmaxf(mloc, __shfl_xor(mloc, d, 32));
      const float mo = mstate[row];
      const float mn = fmaxf(mo, mloc);
      const float al = __expf(mo - mn);
      const float p0 = __expf(sA - mn);
      const float p1 = __expf(sB - mn);
      float rs = p0 + p1;
#pragma unroll
      for (int d = 8; d >= 1; d >>= 1) rs += __shfl_xor(rs, d, 32);
      if (li == 0) {
        mstate[row] = mn;
        lstate[row] = lstate[row] * al + rs;
        alphaS[row] = al;
      }
      Pl[row][li]      = (_Float16)p0;
      Pl[row][li + 16] = (_Float16)p1;
    }
    __syncthreads();

    // O = O*alpha + P @ V  (this wave's 64 channels, 4 WMMA n-tiles)
    float alr[8];
#pragma unroll
    for (int r = 0; r < 8; ++r) alr[r] = alphaS[r + 8 * half];
#pragma unroll
    for (int t = 0; t < 4; ++t)
#pragma unroll
      for (int r = 0; r < 8; ++r) o[t][r] *= alr[r];

    const _Float16* pp = &Pl[li][kA];
    const v16h pf = load_frag(pp, pp + 16);
#pragma unroll
    for (int t = 0; t < 4; ++t) {
      const _Float16* vp = vT +
          (size_t)(b * C_DIM + wave * 64 + t * 16 + li) * N_SPATIAL + j0 + kA;
      o[t] = wmma16(pf, load_frag(vp, vp + 16), o[t]);
    }
    __syncthreads();
  }

  float linv[8];
#pragma unroll
  for (int r = 0; r < 8; ++r) linv[r] = 1.0f / lstate[r + 8 * half];
#pragma unroll
  for (int t = 0; t < 4; ++t) {
    const int col = wave * 64 + t * 16 + li;
#pragma unroll
    for (int r = 0; r < 8; ++r) {
      const int row = i0 + r + 8 * half;
      attn[(size_t)(b * N_SPATIAL + row) * C_DIM + col] = (_Float16)(o[t][r] * linv[r]);
    }
  }
}

// ---- 5) Output projection + bias + residual, fp32 out -----------------------
__global__ __launch_bounds__(128) void proj_kernel(const _Float16* __restrict__ attn,
                                                   const _Float16* __restrict__ woT,
                                                   const float* __restrict__ bo,
                                                   const float* __restrict__ x,
                                                   float* __restrict__ out) {
  const int wave = threadIdx.x >> 5, lane = threadIdx.x & 31;
  const int half = lane >> 4, li = lane & 15, kA = half * 8;
  const int m0 = blockIdx.x * 64 + (wave >> 1) * 32;
  const int n0 = blockIdx.y * 64 + (wave & 1) * 32;

  v8f zero = {};
  v8f acc[2][2];
#pragma unroll
  for (int i = 0; i < 2; ++i)
#pragma unroll
    for (int j = 0; j < 2; ++j) acc[i][j] = zero;

  for (int kb = 0; kb < C_DIM; kb += 32) {
    v16h a[2], bf[2];
#pragma unroll
    for (int i = 0; i < 2; ++i) {
      const _Float16* pa = attn + (size_t)(m0 + i * 16 + li) * C_DIM + kb + kA;
      a[i] = load_frag(pa, pa + 16);
    }
#pragma unroll
    for (int j = 0; j < 2; ++j) {
      const _Float16* pb = woT + (size_t)(n0 + j * 16 + li) * C_DIM + kb + kA;
      bf[j] = load_frag(pb, pb + 16);
    }
#pragma unroll
    for (int i = 0; i < 2; ++i)
#pragma unroll
      for (int j = 0; j < 2; ++j) acc[i][j] = wmma16(a[i], bf[j], acc[i][j]);
  }

#pragma unroll
  for (int j = 0; j < 2; ++j) {
    const int col = n0 + j * 16 + li;
    const float bb = bo[col];
#pragma unroll
    for (int i = 0; i < 2; ++i) {
#pragma unroll
      for (int r = 0; r < 8; ++r) {
        const int row = m0 + i * 16 + r + 8 * half;
        out[(size_t)row * C_DIM + col] =
            acc[i][j][r] + bb + x[(size_t)row * C_DIM + col];
      }
    }
  }
}

// ---- host launch ------------------------------------------------------------
extern "C" void kernel_launch(void* const* d_in, const int* in_sizes, int n_in,
                              void* d_out, int out_size, void* d_ws, size_t ws_size,
                              hipStream_t stream) {
  (void)in_sizes; (void)n_in; (void)out_size; (void)ws_size;
  const float* x     = (const float*)d_in[0];
  const float* gamma = (const float*)d_in[1];
  const float* beta  = (const float*)d_in[2];
  const float* wq = (const float*)d_in[3];
  const float* bq = (const float*)d_in[4];
  const float* wk = (const float*)d_in[5];
  const float* bk = (const float*)d_in[6];
  const float* wv = (const float*)d_in[7];
  const float* bv = (const float*)d_in[8];
  const float* wo = (const float*)d_in[9];
  const float* bo = (const float*)d_in[10];
  float* out = (float*)d_out;

  // workspace layout (f16): wT(2MB) | h/attn(16MB) | q(16MB) | k(16MB) | vT(16MB)
  char* ws = (char*)d_ws;
  _Float16* wT   = (_Float16*)(ws);
  _Float16* hbuf = (_Float16*)(ws + ((size_t)2  << 20));
  _Float16* q    = (_Float16*)(ws + ((size_t)18 << 20));
  _Float16* k    = (_Float16*)(ws + ((size_t)34 << 20));
  _Float16* vT   = (_Float16*)(ws + ((size_t)50 << 20));

  gn_kernel<<<dim3(BATCH * 32), dim3(256), 0, stream>>>(x, gamma, beta, hbuf);
  wconv_kernel<<<dim3(1024, 1, 4), dim3(256), 0, stream>>>(wq, wk, wv, wo, wT);
  qkv_gemm_kernel<<<dim3(NTOT / 64, C_DIM / 64, 3), dim3(128), 0, stream>>>(
      hbuf, wT, bq, bk, bv, q, k, vT);
  attn_kernel<<<dim3(N_SPATIAL / 16, BATCH), dim3(256), 0, stream>>>(q, k, vT, hbuf);
  proj_kernel<<<dim3(NTOT / 64, C_DIM / 64), dim3(128), 0, stream>>>(
      hbuf, wT + (size_t)3 * C_DIM * C_DIM, bo, x, out);
}
